// LayerwiseMergedModelWrapper_7198365188737
// MI455X (gfx1250) — matrix operand order repaired and checked by reading, not compiled, not run on hardware
//
#include <hip/hip_runtime.h>
#include <cstdint>

// ---------------- model constants ----------------
constexpr int Bc  = 2;
constexpr int Sc  = 1024;
constexpr int Dc  = 1024;
constexpr int Hc  = 16;
constexpr int HDc = 64;
constexpr int DFFc= 4096;
constexpr int Vc  = 32000;
constexpr int Lc  = 2;
constexpr int Nc  = Bc * Sc;          // 2048 tokens
constexpr int NTc = (Vc / 256) * 4;   // 500 per-row partials (125 N-tiles x 4 wave columns)

typedef __attribute__((ext_vector_type(16))) __bf16 v16bf;
typedef __attribute__((ext_vector_type(8)))  float  v8f;
typedef __attribute__((ext_vector_type(4)))  int    v4i;

#define AS1 __attribute__((address_space(1)))
#define AS3 __attribute__((address_space(3)))

__device__ inline v8f wmma_bf16(v16bf a, v16bf b, v8f c) {
  return __builtin_amdgcn_wmma_f32_16x16x32_bf16(false, a, false, b, (short)0, c, false, false);
}

// ---- gfx1250 async global->LDS staging (fallback: synchronous b128 copy) ----
#if __has_builtin(__builtin_amdgcn_global_load_async_to_lds_b128)
#define HAVE_ASYNC_LDS 1
#endif

__device__ inline void stage16(const __bf16* g, __bf16* l) {
#ifdef HAVE_ASYNC_LDS
  v4i* gp = (v4i*)g;   // drop const + retype in generic AS
  v4i* lp = (v4i*)l;
  __builtin_amdgcn_global_load_async_to_lds_b128((AS1 v4i*)gp, (AS3 v4i*)lp, 0, 0);
#else
  *(uint4*)l = *(const uint4*)g;
#endif
}
template <int N>
__device__ inline void stage_wait_n() {
#ifdef HAVE_ASYNC_LDS
#if __has_builtin(__builtin_amdgcn_s_wait_asynccnt)
  __builtin_amdgcn_s_wait_asynccnt((short)N);
#else
  asm volatile("s_wait_asynccnt %0" ::"i"(N) : "memory");
#endif
#endif
}

__device__ inline float gelu_tanh(float x) {
  float x3 = x * x * x;
  return 0.5f * x * (1.0f + tanhf(0.7978845608028654f * (x + 0.044715f * x3)));
}

__device__ inline float block_sum256(float v, float* sh) {
  int t = threadIdx.x;
  sh[t] = v; __syncthreads();
  for (int s = 128; s > 0; s >>= 1) { if (t < s) sh[t] += sh[t + s]; __syncthreads(); }
  float r = sh[0]; __syncthreads();
  return r;
}
__device__ inline float block_max256(float v, float* sh) {
  int t = threadIdx.x;
  sh[t] = v; __syncthreads();
  for (int s = 128; s > 0; s >>= 1) { if (t < s) sh[t] = fmaxf(sh[t], sh[t + s]); __syncthreads(); }
  float r = sh[0]; __syncthreads();
  return r;
}

// ---------------- element-wise merge (emb): out_bf16 = base + l0*d0 + l1*d1 ----------------
__global__ void merge_bf16_kernel(const float* __restrict__ base, const float* __restrict__ d,
                                  const float* __restrict__ lam, __bf16* __restrict__ out,
                                  long long n) {
  float l0 = lam[0], l1 = lam[1];
  long long stride = (long long)gridDim.x * blockDim.x;
  for (long long i = (long long)blockIdx.x * blockDim.x + threadIdx.x; i < n; i += stride)
    out[i] = (__bf16)(base[i] + l0 * d[i] + l1 * d[n + i]);
}
__global__ void merge_f32_kernel(const float* __restrict__ base, const float* __restrict__ d,
                                 const float* __restrict__ lam, float* __restrict__ out,
                                 long long n) {
  float l0 = lam[0], l1 = lam[1];
  long long stride = (long long)gridDim.x * blockDim.x;
  for (long long i = (long long)blockIdx.x * blockDim.x + threadIdx.x; i < n; i += stride)
    out[i] = base[i] + l0 * d[i] + l1 * d[n + i];
}

// ---------------- transposing merge: out[l][c][r] = merged(base[l][r][c]), bf16 ----------------
// grid = (C/32, R/32, L); block = 256. sliceN = R*C, fullN = L*R*C (T-major delta stride).
__global__ void merge_bf16_T_kernel(const float* __restrict__ base, const float* __restrict__ dd,
                                    const float* __restrict__ lam, __bf16* __restrict__ out,
                                    int R, int C, long long sliceN, long long fullN) {
  __shared__ float tile[32][33];
  int l = blockIdx.z;
  const float* b0 = base + (size_t)l * sliceN;
  const float* p0 = dd + (size_t)l * sliceN;
  const float* p1 = dd + fullN + (size_t)l * sliceN;
  float l0 = lam[0], l1 = lam[1];
  int r0 = blockIdx.y * 32, c0 = blockIdx.x * 32;
  for (int idx = threadIdx.x; idx < 1024; idx += 256) {
    int r = idx >> 5, c = idx & 31;
    size_t gi = (size_t)(r0 + r) * C + c0 + c;
    tile[r][c] = b0[gi] + l0 * p0[gi] + l1 * p1[gi];
  }
  __syncthreads();
  __bf16* o = out + (size_t)l * sliceN;
  for (int idx = threadIdx.x; idx < 1024; idx += 256) {
    int c = idx >> 5, r = idx & 31;   // write coalesced along r
    o[(size_t)(c0 + c) * R + r0 + r] = (__bf16)tile[r][c];
  }
}

// ---------------- embedding gather (merged on the fly, f32 residual stream) ----------------
__global__ void gather_embed_kernel(const int* __restrict__ ids, const float* __restrict__ emb,
                                    const float* __restrict__ demb, const float* __restrict__ lam,
                                    float* __restrict__ h, long long VD) {
  int row = blockIdx.x;
  int id  = ids[row];
  float l0 = lam[0], l1 = lam[1];
  const float* e  = emb  + (size_t)id * Dc;
  const float* d0 = demb + (size_t)id * Dc;
  const float* d1 = demb + VD + (size_t)id * Dc;
  for (int d = threadIdx.x; d < Dc; d += blockDim.x)
    h[(size_t)row * Dc + d] = e[d] + l0 * d0[d] + l1 * d1[d];
}

// ---------------- LayerNorm (f32 in) -> bf16 out ----------------
__global__ void layernorm_bf16_kernel(const float* __restrict__ x, const float* __restrict__ g,
                                      __bf16* __restrict__ out) {
  __shared__ float sh[256];
  int row = blockIdx.x;
  const float* xr = x + (size_t)row * Dc;
  float s = 0.f, s2 = 0.f;
  for (int d = threadIdx.x; d < Dc; d += 256) { float v = xr[d]; s += v; s2 += v * v; }
  float tot  = block_sum256(s, sh);
  float tot2 = block_sum256(s2, sh);
  float mu  = tot / Dc;
  float var = tot2 / Dc - mu * mu;
  float rs  = rsqrtf(var + 1e-5f);
  for (int d = threadIdx.x; d < Dc; d += 256)
    out[(size_t)row * Dc + d] = (__bf16)((xr[d] - mu) * rs * g[d]);
}

// ================= double-buffered WMMA bf16 GEMM core =================
// C(MxN) = A(MxK) @ Bt^T where Bt is N x K (transposed weights / emb rows).
// Block 256 thr = 8 waves; block tile 64(M) x 256(N); wave tile 32x64 -> 8 WMMA / k-chunk.
// LDS: As[2][64][40], Bs[2][256][40] (transposed; frag reads are contiguous b128).
// Pipeline: prefetch k+1 async, s_wait_asynccnt 5 (A:1 + B:4 copies in flight / thread).

#define GEMM_CORE(APTR, BTPTR, KDIM)                                              \
  const int tid  = threadIdx.x;                                                   \
  const int wave = tid >> 5, lane = tid & 31;                                     \
  const int hl = lane & 15, hi = lane >> 4;                                       \
  const int wm = wave >> 2, wn = wave & 3;                                        \
  const int sam = tid >> 2, sak = (tid & 3) * 8;                                  \
  v8f acc[2][4] = {};                                                             \
  {                                                                               \
    const __bf16* ga = &APTR[(size_t)(m0 + sam) * KDIM + sak];                    \
    stage16(ga, &As[0][sam][sak]);                                                \
    const __bf16* gb = &BTPTR[(size_t)(n0 + tid) * KDIM];                         \
    _Pragma("unroll") for (int j = 0; j < 4; j++)                                 \
        stage16(gb + j * 8, &Bs[0][tid][j * 8]);                                  \
  }                                                                               \
  for (int k0 = 0; k0 < KDIM; k0 += 32) {                                         \
    int cur = (k0 >> 5) & 1;                                                      \
    if (k0 + 32 < KDIM) {                                                         \
      const __bf16* ga = &APTR[(size_t)(m0 + sam) * KDIM + k0 + 32 + sak];        \
      stage16(ga, &As[cur ^ 1][sam][sak]);                                        \
      const __bf16* gb = &BTPTR[(size_t)(n0 + tid) * KDIM + k0 + 32];             \
      _Pragma("unroll") for (int j = 0; j < 4; j++)                               \
          stage16(gb + j * 8, &Bs[cur ^ 1][tid][j * 8]);                          \
      stage_wait_n<5>();                                                          \
    } else {                                                                      \
      stage_wait_n<0>();                                                          \
    }                                                                             \
    __syncthreads();                                                              \
    v16bf a[2], b[4];                                                             \
    _Pragma("unroll") for (int i = 0; i < 2; i++) {                               \
      int am = wm * 32 + i * 16 + hl, ak = hi * 16;                               \
      _Pragma("unroll") for (int q = 0; q < 8; q++) a[i][q] = As[cur][am][ak + q];\
      _Pragma("unroll") for (int q = 0; q < 8; q++)                               \
          a[i][8 + q] = As[cur][am][ak + 8 + q];                                  \
    }                                                                             \
    _Pragma("unroll") for (int j = 0; j < 4; j++) {                               \
      int bn = wn * 64 + j * 16 + hl, bk = hi * 16;                               \
      _Pragma("unroll") for (int q = 0; q < 16; q++) b[j][q] = Bs[cur][bn][bk + q];\
    }                                                                             \
    _Pragma("unroll") for (int i = 0; i < 2; i++)                                 \
      _Pragma("unroll") for (int j = 0; j < 4; j++)                               \
          acc[i][j] = wmma_bf16(a[i], b[j], acc[i][j]);                           \
    __syncthreads();                                                              \
  }

// MODE 0: Cf = r   MODE 1: Cf += r   MODE 2: Cb = bf16(gelu(r))   MODE 3: Cb = bf16(r)
template <int MODE>
__global__ void gemm_bf16_kernel(const __bf16* __restrict__ A, const __bf16* __restrict__ Bt,
                                 float* __restrict__ Cf, __bf16* __restrict__ Cb,
                                 int M, int Nn, int K) {
  __shared__ __bf16 As[2][64][40];
  __shared__ __bf16 Bs[2][256][40];
  const int n0 = blockIdx.x * 256;
  const int m0 = blockIdx.y * 64;
  GEMM_CORE(A, Bt, K)

#pragma unroll
  for (int i = 0; i < 2; i++)
#pragma unroll
    for (int j = 0; j < 4; j++)
#pragma unroll
      for (int r = 0; r < 8; r++) {
        int gm = m0 + wm * 32 + i * 16 + r + hi * 8;
        int gn = n0 + wn * 64 + j * 16 + hl;
        size_t idx = (size_t)gm * Nn + gn;
        if (MODE == 0)      Cf[idx] = acc[i][j][r];
        else if (MODE == 1) Cf[idx] += acc[i][j][r];
        else if (MODE == 2) Cb[idx] = (__bf16)gelu_tanh(acc[i][j][r]);
        else                Cb[idx] = (__bf16)acc[i][j][r];
      }
}

// ---------------- fused vocab GEMM + per-wave-tile (max, sum-exp) partials ----------------
// grid = (V/256, N/64); emb_m is already N x K (row-major [v][d]) -> same fast core.
__global__ void vocab_lse_kernel(const __bf16* __restrict__ X, const __bf16* __restrict__ E,
                                 float* __restrict__ pmax, float* __restrict__ psum) {
  __shared__ __bf16 As[2][64][40];
  __shared__ __bf16 Bs[2][256][40];
  const int n0 = blockIdx.x * 256;
  const int m0 = blockIdx.y * 64;
  GEMM_CORE(X, E, Dc)

  // per-row (max, sum-exp) over this wave's 64 columns; one partial per wave column
#pragma unroll
  for (int i = 0; i < 2; i++) {
#pragma unroll
    for (int r = 0; r < 8; r++) {
      float m = fmaxf(fmaxf(acc[i][0][r], acc[i][1][r]), fmaxf(acc[i][2][r], acc[i][3][r]));
      for (int off = 1; off < 16; off <<= 1) m = fmaxf(m, __shfl_xor(m, off, 16));
      float s = 0.f;
#pragma unroll
      for (int j = 0; j < 4; j++) s += __expf(acc[i][j][r] - m);
      for (int off = 1; off < 16; off <<= 1) s += __shfl_xor(s, off, 16);
      if (hl == 0) {
        int row = m0 + wm * 32 + i * 16 + r + hi * 8;
        size_t pidx = (size_t)row * NTc + (size_t)blockIdx.x * 4 + wn;
        pmax[pidx] = m;
        psum[pidx] = s;
      }
    }
  }
}

// ---------------- flash-style causal attention, all WMMA (qkv bf16) ----------------
// grid = (B*H, S/128), block = 256 (8 waves, 16 q-rows each).
__global__ void attention_kernel(const __bf16* __restrict__ qkv, __bf16* __restrict__ obf) {
  const int bh = blockIdx.x;
  const int b  = bh / Hc, hh = bh % Hc;
  const int qbase = blockIdx.y * 128;
  const int tid = threadIdx.x;
  const int wave = tid >> 5, lane = tid & 31;
  const int hl = lane & 15, hi = lane >> 4;
  const int qrow0 = qbase + wave * 16;

  __shared__ __bf16 Ks[32][HDc + 8];
  __shared__ __bf16 Vs[32][HDc + 8];
  __shared__ __bf16 Ps[8][16][32];

  v16bf aQ[2];
  {
    size_t tok = (size_t)b * Sc + qrow0 + hl;
    const __bf16* qp = qkv + tok * (3 * Dc) + hh * HDc;
    int kof = hi * 16;
#pragma unroll
    for (int c = 0; c < 2; c++) {
#pragma unroll
      for (int i = 0; i < 8; i++) aQ[c][i]     = qp[c * 32 + kof + i];
#pragma unroll
      for (int i = 0; i < 8; i++) aQ[c][8 + i] = qp[c * 32 + kof + 8 + i];
    }
  }

  v8f acc[4] = {};
  float m8[8], l8[8];
#pragma unroll
  for (int r = 0; r < 8; r++) { m8[r] = -3.0e38f; l8[r] = 0.f; }

  const int kend = qbase + 128;
  for (int kb = 0; kb < kend; kb += 32) {
    {
      int kr = tid >> 3, d8 = (tid & 7) * 8;
      size_t tok = (size_t)b * Sc + kb + kr;
      const __bf16* base = qkv + tok * (3 * Dc) + hh * HDc;
      stage16(base + Dc,     &Ks[kr][d8]);
      stage16(base + 2 * Dc, &Vs[kr][d8]);
    }
    stage_wait_n<0>();
    __syncthreads();

    v8f c0 = {}, c1 = {};
#pragma unroll
    for (int dc = 0; dc < 2; dc++) {
      v16bf bk0, bk1;
#pragma unroll
      for (int i = 0; i < 16; i++) bk0[i] = Ks[hl][dc * 32 + hi * 16 + i];
#pragma unroll
      for (int i = 0; i < 16; i++) bk1[i] = Ks[16 + hl][dc * 32 + hi * 16 + i];
      c0 = wmma_bf16(aQ[dc], bk0, c0);
      c1 = wmma_bf16(aQ[dc], bk1, c1);
    }

#pragma unroll
    for (int r = 0; r < 8; r++) {
      int qa  = qrow0 + r + hi * 8;
      int k0c = kb + hl, k1c = kb + 16 + hl;
      float s0 = (k0c <= qa) ? c0[r] * 0.125f : -1.0e9f;
      float s1 = (k1c <= qa) ? c1[r] * 0.125f : -1.0e9f;
      float tm = fmaxf(s0, s1);
      for (int off = 1; off < 16; off <<= 1) tm = fmaxf(tm, __shfl_xor(tm, off, 16));
      float mnew = fmaxf(m8[r], tm);
      float corr = __expf(m8[r] - mnew);
      float p0 = __expf(s0 - mnew), p1 = __expf(s1 - mnew);
      float rs = p0 + p1;
      for (int off = 1; off < 16; off <<= 1) rs += __shfl_xor(rs, off, 16);
      l8[r] = l8[r] * corr + rs;
      m8[r] = mnew;
#pragma unroll
      for (int f = 0; f < 4; f++) acc[f][r] *= corr;
      int ml = r + hi * 8;
      Ps[wave][ml][hl]      = (__bf16)p0;
      Ps[wave][ml][16 + hl] = (__bf16)p1;
    }

    v16bf aP;
    {
      int kof = hi * 16;
#pragma unroll
      for (int i = 0; i < 8; i++) aP[i]     = Ps[wave][hl][kof + i];
#pragma unroll
      for (int i = 0; i < 8; i++) aP[8 + i] = Ps[wave][hl][kof + 8 + i];
    }
#pragma unroll
    for (int f = 0; f < 4; f++) {
      v16bf bv;
#pragma unroll
      for (int i = 0; i < 16; i++) bv[i] = Vs[hi * 16 + i][f * 16 + hl];
      acc[f] = wmma_bf16(aP, bv, acc[f]);
    }
    __syncthreads();
  }

#pragma unroll
  for (int r = 0; r < 8; r++) {
    float inv = 1.0f / l8[r];
    int qa = qrow0 + r + hi * 8;
    size_t tok = (size_t)b * Sc + qa;
#pragma unroll
    for (int f = 0; f < 4; f++)
      obf[tok * Dc + hh * HDc + f * 16 + hl] = (__bf16)(acc[f][r] * inv);
  }
}

// ---------------- target-logit gather-dot ----------------
__global__ void target_logit_kernel(const __bf16* __restrict__ X, const __bf16* __restrict__ E,
                                    const int* __restrict__ labels, float* __restrict__ tl) {
  __shared__ float sh[256];
  int row = blockIdx.x;
  int b = row / Sc, s = row % Sc;
  if (s == Sc - 1) { if (threadIdx.x == 0) tl[row] = 0.f; return; }
  int tgt = labels[b * Sc + s + 1];
  float a = 0.f;
  for (int d = threadIdx.x; d < Dc; d += 256)
    a += (float)X[(size_t)row * Dc + d] * (float)E[(size_t)tgt * Dc + d];
  float tot = block_sum256(a, sh);
  if (threadIdx.x == 0) tl[row] = tot;
}

// ---------------- logsumexp combine + mean NLL ----------------
__global__ void init_accum_kernel(float* a) { a[0] = 0.f; }
__global__ void nll_kernel(const float* __restrict__ pmax, const float* __restrict__ psum,
                           const float* __restrict__ tl, float* __restrict__ accum) {
  __shared__ float sh[256];
  int row = blockIdx.x;
  int s = row % Sc;
  if (s == Sc - 1) return;
  float lm = -3.0e38f;
  for (int t = threadIdx.x; t < NTc; t += 256)
    lm = fmaxf(lm, pmax[(size_t)row * NTc + t]);
  float M = block_max256(lm, sh);
  float ls = 0.f;
  for (int t = threadIdx.x; t < NTc; t += 256)
    ls += psum[(size_t)row * NTc + t] * __expf(pmax[(size_t)row * NTc + t] - M);
  float Ssum = block_sum256(ls, sh);
  if (threadIdx.x == 0) {
    float lse = M + __logf(Ssum);
    atomicAdd(accum, lse - tl[row]);
  }
}
__global__ void finalize_kernel(const float* accum, float* out) {
  out[0] = accum[0] * (1.0f / (float)(Bc * (Sc - 1)));
}

// ---------------- host orchestration ----------------
extern "C" void kernel_launch(void* const* d_in, const int* in_sizes, int n_in,
                              void* d_out, int out_size, void* d_ws, size_t ws_size,
                              hipStream_t stream) {
  (void)in_sizes; (void)n_in; (void)out_size; (void)ws_size;
  const int*   ids    = (const int*)d_in[0];
  const int*   labels = (const int*)d_in[1];
  const float* lam    = (const float*)d_in[2];
  const float* emb    = (const float*)d_in[3];
  const float* Wqkv   = (const float*)d_in[4];
  const float* Wo     = (const float*)d_in[5];
  const float* W1     = (const float*)d_in[6];
  const float* W2     = (const float*)d_in[7];
  const float* g1     = (const float*)d_in[8];
  const float* g2     = (const float*)d_in[9];
  const float* gf     = (const float*)d_in[10];
  const float* d_emb  = (const float*)d_in[11];
  const float* d_Wqkv = (const float*)d_in[12];
  const float* d_Wo   = (const float*)d_in[13];
  const float* d_W1   = (const float*)d_in[14];
  const float* d_W2   = (const float*)d_in[15];
  const float* d_g1   = (const float*)d_in[16];
  const float* d_g2   = (const float*)d_in[17];
  const float* d_gf   = (const float*)d_in[18];
  float* out = (float*)d_out;

  char* ws = (char*)d_ws;
  size_t off = 0;
  auto carve = [&](size_t bytes) -> char* {
    off = (off + 255) & ~(size_t)255;
    char* p = ws + off; off += bytes; return p;
  };

  const long long nEmb = (long long)Vc * Dc;
  const long long nQkv = (long long)Lc * Dc * 3 * Dc;
  const long long nWo  = (long long)Lc * Dc * Dc;
  const long long nW1  = (long long)Lc * Dc * DFFc;
  const long long nW2  = (long long)Lc * DFFc * Dc;

  __bf16* emb_bf  = (__bf16*)carve((size_t)nEmb * 2);     // [V][D] (= B^T for vocab GEMM)
  __bf16* wqkvT   = (__bf16*)carve((size_t)nQkv * 2);     // [L][3D][D]
  __bf16* woT     = (__bf16*)carve((size_t)nWo  * 2);     // [L][D][D]
  __bf16* w1T     = (__bf16*)carve((size_t)nW1  * 2);     // [L][DFF][D]
  __bf16* w2T     = (__bf16*)carve((size_t)nW2  * 2);     // [L][D][DFF]
  float*  g1m     = (float*) carve((size_t)Lc * Dc * 4);
  float*  g2m     = (float*) carve((size_t)Lc * Dc * 4);
  float*  gfm     = (float*) carve((size_t)Dc * 4);
  float*  h       = (float*) carve((size_t)Nc * Dc * 4);
  __bf16* a_bf    = (__bf16*)carve((size_t)Nc * Dc * 2);
  __bf16* qkv_bf  = (__bf16*)carve((size_t)Nc * 3 * Dc * 2);
  __bf16* o_bf    = (__bf16*)carve((size_t)Nc * Dc * 2);
  __bf16* u_bf    = (__bf16*)carve((size_t)Nc * DFFc * 2);
  __bf16* xf_bf   = (__bf16*)carve((size_t)Nc * Dc * 2);
  float*  pmax    = (float*) carve((size_t)Nc * NTc * 4);
  float*  psum    = (float*) carve((size_t)Nc * NTc * 4);
  float*  tl      = (float*) carve((size_t)Nc * 4);
  float*  accum   = (float*) carve(4);

  // --- weight merges (stream once; matmul weights stored transposed bf16) ---
  merge_bf16_kernel<<<dim3(8192), dim3(256), 0, stream>>>(emb, d_emb, lam, emb_bf, nEmb);
  merge_bf16_T_kernel<<<dim3(3 * Dc / 32, Dc / 32, Lc), dim3(256), 0, stream>>>(
      Wqkv, d_Wqkv, lam, wqkvT, Dc, 3 * Dc, (long long)Dc * 3 * Dc, nQkv);
  merge_bf16_T_kernel<<<dim3(Dc / 32, Dc / 32, Lc), dim3(256), 0, stream>>>(
      Wo, d_Wo, lam, woT, Dc, Dc, (long long)Dc * Dc, nWo);
  merge_bf16_T_kernel<<<dim3(DFFc / 32, Dc / 32, Lc), dim3(256), 0, stream>>>(
      W1, d_W1, lam, w1T, Dc, DFFc, (long long)Dc * DFFc, nW1);
  merge_bf16_T_kernel<<<dim3(Dc / 32, DFFc / 32, Lc), dim3(256), 0, stream>>>(
      W2, d_W2, lam, w2T, DFFc, Dc, (long long)DFFc * Dc, nW2);
  merge_f32_kernel<<<dim3(8), dim3(256), 0, stream>>>(g1, d_g1, lam, g1m, (long long)Lc * Dc);
  merge_f32_kernel<<<dim3(8), dim3(256), 0, stream>>>(g2, d_g2, lam, g2m, (long long)Lc * Dc);
  merge_f32_kernel<<<dim3(4), dim3(256), 0, stream>>>(gf, d_gf, lam, gfm, (long long)Dc);

  // --- embedding gather ---
  gather_embed_kernel<<<dim3(Nc), dim3(256), 0, stream>>>(ids, emb, d_emb, lam, h, nEmb);

  // --- transformer layers ---
  for (int l = 0; l < Lc; l++) {
    layernorm_bf16_kernel<<<dim3(Nc), dim3(256), 0, stream>>>(h, g1m + l * Dc, a_bf);
    gemm_bf16_kernel<3><<<dim3(3 * Dc / 256, Nc / 64), dim3(256), 0, stream>>>(
        a_bf, wqkvT + (size_t)l * Dc * 3 * Dc, nullptr, qkv_bf, Nc, 3 * Dc, Dc);
    attention_kernel<<<dim3(Bc * Hc, Sc / 128), dim3(256), 0, stream>>>(qkv_bf, o_bf);
    gemm_bf16_kernel<1><<<dim3(Dc / 256, Nc / 64), dim3(256), 0, stream>>>(
        o_bf, woT + (size_t)l * Dc * Dc, h, nullptr, Nc, Dc, Dc);
    layernorm_bf16_kernel<<<dim3(Nc), dim3(256), 0, stream>>>(h, g2m + l * Dc, a_bf);
    gemm_bf16_kernel<2><<<dim3(DFFc / 256, Nc / 64), dim3(256), 0, stream>>>(
        a_bf, w1T + (size_t)l * Dc * DFFc, nullptr, u_bf, Nc, DFFc, Dc);
    gemm_bf16_kernel<1><<<dim3(Dc / 256, Nc / 64), dim3(256), 0, stream>>>(
        u_bf, w2T + (size_t)l * DFFc * Dc, h, nullptr, Nc, Dc, DFFc);
  }

  // --- final LN + fused vocab logsumexp + NLL ---
  layernorm_bf16_kernel<<<dim3(Nc), dim3(256), 0, stream>>>(h, gfm, xf_bf);
  vocab_lse_kernel<<<dim3(Vc / 256, Nc / 64), dim3(256), 0, stream>>>(xf_bf, emb_bf, pmax, psum);
  target_logit_kernel<<<dim3(Nc), dim3(256), 0, stream>>>(xf_bf, emb_bf, labels, tl);
  init_accum_kernel<<<dim3(1), dim3(1), 0, stream>>>(accum);
  nll_kernel<<<dim3(Nc), dim3(256), 0, stream>>>(pmax, psum, tl, accum);
  finalize_kernel<<<dim3(1), dim3(1), 0, stream>>>(accum, out);
}
